// FixedPointFasterKANLayer_44908178047569
// MI455X (gfx1250) — 8-bit matrix operand order repaired and checked by reading, compiled, not run on hardware
//
#include <hip/hip_runtime.h>
#include <hip/hip_bf16.h>


typedef __attribute__((ext_vector_type(8))) int v8i;

static constexpr int Bsz = 8192;
static constexpr int Dsz = 1024;
static constexpr int Gsz = 8;
static constexpr int Osz = 1024;
static constexpr int Ksz = Dsz * Gsz;

static constexpr int BM = 128;
static constexpr int BN = 128;
static constexpr int BK = 128;

#if __has_builtin(__builtin_amdgcn_global_load_async_to_lds_b128)
#define HAVE_ASYNC_LDS 1
typedef int b128_t __attribute__((vector_size(16)));
typedef __attribute__((address_space(1))) b128_t* gbl_b128_p;
typedef __attribute__((address_space(3))) b128_t* lds_b128_p;

__device__ __forceinline__ void async_copy16(const void* g, void* l) {
    __builtin_amdgcn_global_load_async_to_lds_b128(
        (gbl_b128_p)(void*)g, (lds_b128_p)l,   0,   0);
}
__device__ __forceinline__ void wait_async0() {
#if __has_builtin(__builtin_amdgcn_s_wait_asynccnt)
    __builtin_amdgcn_s_wait_asynccnt(0);
#else
    asm volatile("s_wait_asynccnt 0x0" ::: "memory");
#endif
}
#else
#define HAVE_ASYNC_LDS 0
#endif

__global__ __launch_bounds__(256) void kan_act_kernel(
    const float* __restrict__ x, const int* __restrict__ grid8,
    const int* __restrict__ inv_denom, unsigned char* __restrict__ Aq)
{
    int idx = blockIdx.x * blockDim.x + threadIdx.x;
    float xf = x[idx];
    int xq = (int)rintf(xf * 16.0f);
    xq = xq < -128 ? -128 : (xq > 127 ? 127 : xq);
    int scale = inv_denom[0];

    unsigned int lo = 0u, hi = 0u;
#pragma unroll
    for (int g = 0; g < Gsz; ++g) {
        int d8   = (int)(signed char)(xq - grid8[g]);
        int prod = d8 * scale;
        int sd   = (int)(signed char)(prod >> 4);
        float s  = (float)sd * 0.0625f;
        float t  = tanhf(s);
        float act = 1.0f - t * t;
        int q = (int)rintf(act * 128.0f);
        q = q < 0 ? 0 : (q > 127 ? 127 : q);
        if (g < 4) lo |= (unsigned)q << (8 * g);
        else       hi |= (unsigned)q << (8 * (g - 4));
    }
    ((uint2*)Aq)[idx] = make_uint2(lo, hi);
}

__global__ __launch_bounds__(256) void kan_pack_w_kernel(
    const int* __restrict__ w, unsigned char* __restrict__ wp)
{
    int idx = blockIdx.x * blockDim.x + threadIdx.x;
    int4 v = ((const int4*)w)[idx];
    unsigned int p = ((unsigned)v.x & 0xffu)
                   | (((unsigned)v.y & 0xffu) << 8)
                   | (((unsigned)v.z & 0xffu) << 16)
                   | (((unsigned)v.w & 0xffu) << 24);
    ((unsigned int*)wp)[idx] = p;
}

__global__ __launch_bounds__(256) void kan_gemm_kernel(
    const unsigned char* __restrict__ A,
    const unsigned char* __restrict__ Wp,
    float* __restrict__ out)
{
    __shared__ __align__(16) unsigned char lA[2][BM * BK];
    __shared__ __align__(16) unsigned char lB[2][BN * BK];

    const int tid  = threadIdx.x;
    const int lane = tid & 31;
    const int wv   = tid >> 5;
    const int wm   = wv & 1;
    const int wn   = wv >> 1;
    const int m0   = blockIdx.y * BM;
    const int n0   = blockIdx.x * BN;
    const int l15  = lane & 15;
    const int lhi  = lane >> 4;

    v8i c[4][2];
#pragma unroll
    for (int mi = 0; mi < 4; ++mi)
#pragma unroll
        for (int ni = 0; ni < 2; ++ni)
            c[mi][ni] = (v8i){0, 0, 0, 0, 0, 0, 0, 0};

    auto load_tiles = [&](int kt, int buf) {
#pragma unroll
        for (int i = 0; i < 4; ++i) {
            int li  = tid + i * 256;
            int row = li >> 3;
            int col = (li & 7) << 4;
            const unsigned char* ga = A  + (size_t)(m0 + row) * Ksz + kt + col;
            const unsigned char* gb = Wp + (size_t)(n0 + row) * Ksz + kt + col;
#if HAVE_ASYNC_LDS
            async_copy16(ga, &lA[buf][row * BK + col]);
            async_copy16(gb, &lB[buf][row * BK + col]);
#else
            *(uint4*)(&lA[buf][row * BK + col]) = *(const uint4*)ga;
            *(uint4*)(&lB[buf][row * BK + col]) = *(const uint4*)gb;
#endif
        }
    };

    load_tiles(0, 0);
#if HAVE_ASYNC_LDS
    wait_async0();
#endif
    __syncthreads();

    for (int kt = 0; kt < Ksz; kt += BK) {
        const int buf = (kt / BK) & 1;
        if (kt + BK < Ksz) load_tiles(kt + BK, buf ^ 1);

#pragma unroll
        for (int kk = 0; kk < BK; kk += 64) {
            v8i bf[2];
#pragma unroll
            for (int ni = 0; ni < 2; ++ni) {
                int col = wn * 32 + ni * 16 + l15;
                const unsigned char* p = &lB[buf][col * BK + kk + (lhi << 4)];
                uint4 q0 = *(const uint4*)(p);
                uint4 q1 = *(const uint4*)(p + 32);
                bf[ni] = (v8i){(int)q0.x, (int)q0.y, (int)q0.z, (int)q0.w,
                               (int)q1.x, (int)q1.y, (int)q1.z, (int)q1.w};
            }
            v8i af[4];
#pragma unroll
            for (int mi = 0; mi < 4; ++mi) {
                int row = wm * 64 + mi * 16 + l15;
                const unsigned char* p = &lA[buf][row * BK + kk + (lhi << 4)];
                uint4 a0 = *(const uint4*)(p);
                uint4 a1 = *(const uint4*)(p + 32);
                af[mi] = (v8i){(int)a0.x, (int)a0.y, (int)a0.z, (int)a0.w,
                               (int)a1.x, (int)a1.y, (int)a1.z, (int)a1.w};
            }
#pragma unroll
            for (int mi = 0; mi < 4; ++mi) {
#pragma unroll
                for (int ni = 0; ni < 2; ++ni) {
                    c[mi][ni] = __builtin_amdgcn_wmma_i32_16x16x64_iu8(
                        false, af[mi], true, bf[ni], c[mi][ni], false, false);
                }
            }
        }
#if HAVE_ASYNC_LDS
        wait_async0();
#endif
        __syncthreads();
    }

    const int mrow = m0 + wm * 64 + (lhi << 3);
    const int ncol = n0 + wn * 32 + l15;
#pragma unroll
    for (int mi = 0; mi < 4; ++mi) {
#pragma unroll
        for (int ni = 0; ni < 2; ++ni) {
#pragma unroll
            for (int r = 0; r < 8; ++r) {
                int row = mrow + mi * 16 + r;
                int col = ncol + ni * 16;
                int acc = c[mi][ni][r];
                signed char q = (signed char)(acc >> 7);
                out[(size_t)row * Osz + col] = (float)q;
            }
        }
    }
}

extern "C" void kernel_launch(void* const* d_in, const int* in_sizes, int n_in,
                              void* d_out, int out_size, void* d_ws, size_t ws_size,
                              hipStream_t stream) {
    const float* x      = (const float*)d_in[0];
    const int*   grid8  = (const int*)d_in[1];
    const int*   invd   = (const int*)d_in[2];
    const int*   weight = (const int*)d_in[3];
    float*       out    = (float*)d_out;

    unsigned char* Aq = (unsigned char*)d_ws;
    unsigned char* Wp = Aq + (size_t)Bsz * Ksz;

    (void)in_sizes; (void)n_in; (void)out_size; (void)ws_size;

    kan_act_kernel<<<(Bsz * Dsz) / 256, 256, 0, stream>>>(x, grid8, invd, Aq);
    kan_pack_w_kernel<<<((size_t)Osz * Ksz / 4) / 256, 256, 0, stream>>>(weight, Wp);

    dim3 g(Osz / BN, Bsz / BM);
    kan_gemm_kernel<<<g, 256, 0, stream>>>(Aq, Wp, out);
}
